// CausalLinearAttention_31215822307442
// MI455X (gfx1250) — matrix operand order repaired and checked by reading, not compiled, not run on hardware
//
#include <hip/hip_runtime.h>

// ---------------------------------------------------------------------------
// Causal linear attention (chunked), MI455X / gfx1250, wave32 + WMMA bf16.
//   1) fp32 -> bf16 casts (x, Wq, Wk, Wv, Wo)
//   2) q/k/v projections as bf16 WMMA GEMMs, elu+1 fused for q,k
//   3) per-chunk KV = phi_k^T @ v and Ksum (WMMA; async-to-LDS staging)
//   4) exclusive prefix scan over chunks -> S_prefix, z_prefix (in place)
//   5) per-chunk attention (WMMA; v chunk staged via Tensor Data Mover)
//   6) out = y @ Wo^T as WMMA GEMM, fp32 epilogue
// ---------------------------------------------------------------------------

#define BATCH   4
#define SEQ     4096
#define D_MODEL 1024
#define N_HEADS 16
#define DK      64
#define CHUNK   64
#define NCHUNK  (SEQ / CHUNK)
#define MTOK    (BATCH * SEQ)

typedef __attribute__((ext_vector_type(16))) __bf16        v16bf;
typedef __attribute__((ext_vector_type(8)))  float         v8f;
typedef __attribute__((ext_vector_type(4)))  unsigned int  u32x4;
typedef __attribute__((ext_vector_type(4)))  unsigned int  su32x4;
typedef __attribute__((ext_vector_type(8)))  unsigned int  su32x8;

#define DEVINL static __device__ __forceinline__

union AFrag { v16bf bf; u32x4 u4[2]; unsigned short s[16]; };
union CFrag { v8f v; float f[8]; };

DEVINL unsigned short f2bf(float f) {
    union { float f; unsigned u; } a; a.f = f;
    unsigned u = a.u + 0x7FFFu + ((a.u >> 16) & 1u);   // round-to-nearest-even
    return (unsigned short)(u >> 16);
}
DEVINL float bf2f(unsigned short h) {
    union { unsigned u; float f; } a; a.u = ((unsigned)h) << 16;
    return a.f;
}
DEVINL float elu1(float x) { return x >= 0.f ? x + 1.f : __expf(x); }

DEVINL v8f wmma_bf16(v16bf a, v16bf b, v8f c) {
    return __builtin_amdgcn_wmma_f32_16x16x32_bf16(
        false, a, false, b, (short)0, c, false, false);
}

// Async memory->LDS (ASYNCcnt). ldsOff: per-lane LDS byte address.
DEVINL void async_ld_b128(unsigned ldsOff, const void* g) {
    unsigned long long ga = (unsigned long long)g;
    asm volatile("global_load_async_to_lds_b128 %0, %1, off"
                 :: "v"(ldsOff), "v"(ga) : "memory");
}
DEVINL void wait_asynccnt0() {
    asm volatile("s_wait_asynccnt 0x0" ::: "memory");
}

// Tensor Data Mover: 2D tile of bf16 (rows x cols, row stride in elements)
// from global memory to contiguous LDS at ldsOff. One issue per wave.
DEVINL void tdm_load_tile_bf16(unsigned ldsOff, const void* g,
                               unsigned rows, unsigned cols,
                               unsigned strideElems) {
    unsigned long long ga = (unsigned long long)g;
    union { su32x4 v; unsigned w[4]; } G0;
    G0.w[0] = 1u;                                   // count=1 user descriptor
    G0.w[1] = ldsOff;                               // lds_addr
    G0.w[2] = (unsigned)ga;                         // global_addr[31:0]
    G0.w[3] = (unsigned)((ga >> 32) & 0x01FFFFFFu)  // global_addr[56:32]
            | (2u << 30);                           // type=2 (image)
    union { su32x8 v; unsigned w[8]; } G1;
    G1.w[0] = 1u << 16;                             // data_size=1 (2 bytes)
    G1.w[1] = (cols & 0xFFFFu) << 16;               // tensor_dim0[15:0]
    G1.w[2] = ((cols >> 16) & 0xFFFFu)              // tensor_dim0[31:16]
            | ((rows & 0xFFFFu) << 16);             // tensor_dim1[15:0]
    G1.w[3] = ((rows >> 16) & 0xFFFFu)              // tensor_dim1[31:16]
            | ((cols & 0xFFFFu) << 16);             // tile_dim0
    G1.w[4] = rows & 0xFFFFu;                       // tile_dim1 (tile_dim2=0)
    G1.w[5] = strideElems;                          // tensor_dim0_stride[31:0]
    G1.w[6] = 0u;                                   // stride0 hi / stride1 lo
    G1.w[7] = 0u;
    asm volatile("tensor_load_to_lds %0, %1" :: "s"(G0.v), "s"(G1.v) : "memory");
}

// ---------------------------------------------------------------------------
// 1) fp32 -> bf16 cast
// ---------------------------------------------------------------------------
__global__ void k_cast_bf16(const float* __restrict__ in,
                            unsigned short* __restrict__ out, int n) {
    int i = blockIdx.x * blockDim.x + threadIdx.x;
    if (i < n) out[i] = f2bf(in[i]);
}

// ---------------------------------------------------------------------------
// 2/6) GEMM: out[m,n] = sum_k A[m,k] * W[n,k]   (A: MxK bf16, W: NxK bf16)
// 256 thr = 8 waves (4x2); block tile 128x128; wave 32x64; K step 32.
// mode 0: elu+1 -> bf16 ; mode 1: -> bf16 ; mode 2: -> fp32
// ---------------------------------------------------------------------------
__global__ void __launch_bounds__(256)
k_gemm_nt(const unsigned short* __restrict__ A,
          const unsigned short* __restrict__ Bw,
          unsigned short* __restrict__ outBf,
          float* __restrict__ outF,
          int M, int N, int K, int mode) {
    const int lane = threadIdx.x & 31;
    const int wave = threadIdx.x >> 5;
    const int half = lane >> 4;
    const int r    = lane & 15;
    const int m0   = blockIdx.x * 128 + (wave >> 1) * 32;
    const int n0   = blockIdx.y * 128 + (wave & 1) * 64;

    CFrag acc[2][4];
#pragma unroll
    for (int mi = 0; mi < 2; ++mi)
#pragma unroll
        for (int ni = 0; ni < 4; ++ni)
#pragma unroll
            for (int e = 0; e < 8; ++e) acc[mi][ni].f[e] = 0.f;

    for (int k0 = 0; k0 < K; k0 += 32) {
        __builtin_prefetch(A + (size_t)(m0 + r) * K + k0 + 128, 0, 0);
        AFrag af[2];
#pragma unroll
        for (int mi = 0; mi < 2; ++mi) {
            const unsigned short* p =
                A + (size_t)(m0 + mi * 16 + r) * K + k0 + half * 16;
            af[mi].u4[0] = *(const u32x4*)p;
            af[mi].u4[1] = *(const u32x4*)(p + 8);
        }
        AFrag bfr[4];
#pragma unroll
        for (int ni = 0; ni < 4; ++ni) {
            const unsigned short* p =
                Bw + (size_t)(n0 + ni * 16 + r) * K + k0 + half * 16;
            bfr[ni].u4[0] = *(const u32x4*)p;
            bfr[ni].u4[1] = *(const u32x4*)(p + 8);
        }
#pragma unroll
        for (int mi = 0; mi < 2; ++mi)
#pragma unroll
            for (int ni = 0; ni < 4; ++ni)
                acc[mi][ni].v = wmma_bf16(af[mi].bf, bfr[ni].bf, acc[mi][ni].v);
    }

#pragma unroll
    for (int mi = 0; mi < 2; ++mi)
#pragma unroll
        for (int ni = 0; ni < 4; ++ni)
#pragma unroll
            for (int j = 0; j < 8; ++j) {
                int row = m0 + mi * 16 + half * 8 + j;
                int col = n0 + ni * 16 + r;
                float f = acc[mi][ni].f[j];
                if (mode == 0) f = elu1(f);
                if (mode == 2) outF[(size_t)row * N + col] = f;
                else           outBf[(size_t)row * N + col] = f2bf(f);
            }
}

// ---------------------------------------------------------------------------
// 3) Per-chunk KV = phi_k^T @ v (64x64 per (b,n,h)) and Ksum.
// Block 128 thr = 4 waves; one head per wave.
// Staging via GLOBAL_LOAD_ASYNC_TO_LDS_B128 (ASYNCcnt).
// ---------------------------------------------------------------------------
__global__ void __launch_bounds__(128)
k_chunk_kv(const unsigned short* __restrict__ phiK,
           const unsigned short* __restrict__ Vb,
           float* __restrict__ KV,      // [B*N*H][64*64]
           float* __restrict__ Ksum) {  // [B*N*H][64]
    __shared__ unsigned short lds[4][2][CHUNK * DK];   // LDS base offset 0
    const int lane = threadIdx.x & 31;
    const int wave = threadIdx.x >> 5;
    const int half = lane >> 4;
    const int r    = lane & 15;
    const int hgrp = blockIdx.x & 3;
    const int bn   = blockIdx.x >> 2;
    const int n    = bn % NCHUNK;
    const int b    = bn / NCHUNK;
    const int h    = hgrp * 4 + wave;

    unsigned short* lk = &lds[wave][0][0];
    unsigned short* lv = &lds[wave][1][0];
    const unsigned ldsBaseK = (unsigned)(wave * 2) * (CHUNK * DK * 2);
    const unsigned ldsBaseV = ldsBaseK + CHUNK * DK * 2;

    const size_t cbase = ((size_t)(b * SEQ + n * CHUNK)) * D_MODEL + h * DK;
#pragma unroll
    for (int it = 0; it < 16; ++it) {
        int idx = lane + it * 32;       // 0..511 16B segments
        int row = idx >> 3, seg = idx & 7;
        size_t g = cbase + (size_t)row * D_MODEL + seg * 8;
        async_ld_b128(ldsBaseK + idx * 16, phiK + g);
        async_ld_b128(ldsBaseV + idx * 16, Vb + g);
    }
    wait_asynccnt0();
    asm volatile("s_wait_dscnt 0" ::: "memory");

    // Ksum: column sums of phi_k
#pragma unroll
    for (int t = 0; t < 2; ++t) {
        int k = lane + t * 32;
        float s = 0.f;
        for (int row = 0; row < CHUNK; ++row) s += bf2f(lk[row * DK + k]);
        Ksum[((size_t)(b * NCHUNK + n) * N_HEADS + h) * DK + k] = s;
    }

    float* kvout = KV + ((size_t)(b * NCHUNK + n) * N_HEADS + h) * (DK * DK);
    for (int mt = 0; mt < 4; ++mt)
        for (int nt = 0; nt < 4; ++nt) {
            CFrag acc;
#pragma unroll
            for (int e = 0; e < 8; ++e) acc.f[e] = 0.f;
#pragma unroll
            for (int cs = 0; cs < 2; ++cs) {
                int c0 = cs * 32;
                AFrag a, bb;
#pragma unroll
                for (int i = 0; i < 8; ++i) {
                    a.s[i]     = lk[(c0 + half * 16 + i) * DK + mt * 16 + r];
                    a.s[8 + i] = lk[(c0 + half * 16 + 8 + i) * DK + mt * 16 + r];
                }
#pragma unroll
                for (int i = 0; i < 16; ++i)
                    bb.s[i] = lv[(c0 + half * 16 + i) * DK + nt * 16 + r];
                acc.v = wmma_bf16(a.bf, bb.bf, acc.v);
            }
#pragma unroll
            for (int j = 0; j < 8; ++j)
                kvout[(mt * 16 + half * 8 + j) * DK + nt * 16 + r] = acc.f[j];
        }
}

// ---------------------------------------------------------------------------
// 4) In-place exclusive prefix scan over the chunk axis. [B][N][H][E]
// ---------------------------------------------------------------------------
__global__ void k_excl_scan(float* __restrict__ data, int E) {
    int idx = blockIdx.x * blockDim.x + threadIdx.x;
    int total = BATCH * N_HEADS * E;
    if (idx >= total) return;
    int e = idx % E;
    int h = (idx / E) % N_HEADS;
    int b = idx / (E * N_HEADS);
    float run = 0.f;
    for (int n = 0; n < NCHUNK; ++n) {
        size_t off = (((size_t)b * NCHUNK + n) * N_HEADS + h) * (size_t)E + e;
        float t = data[off];
        data[off] = run;
        run += t;
    }
}

// ---------------------------------------------------------------------------
// 5) Per-chunk attention. Block 64 thr = 2 waves; one (b,n,h) per wave.
// v chunk staged with the Tensor Data Mover (TENSORcnt).
// ---------------------------------------------------------------------------
struct alignas(16) WaveLds {
    unsigned short sbf[DK * DK];        // S_prefix as bf16
    unsigned short attn[CHUNK * CHUNK];
    unsigned short vbf[CHUNK * DK];
    float zrow[CHUNK];
};

__global__ void __launch_bounds__(64)
k_attn(const unsigned short* __restrict__ phiQ,
       const unsigned short* __restrict__ phiK,
       const unsigned short* __restrict__ Vb,
       const float* __restrict__ Spref,    // scanned KV
       const float* __restrict__ zpref,    // scanned Ksum
       unsigned short* __restrict__ yout) {
    __shared__ WaveLds lw[2];               // LDS base offset 0
    const int lane = threadIdx.x & 31;
    const int wave = threadIdx.x >> 5;
    const int half = lane >> 4;
    const int r    = lane & 15;
    const int hp   = blockIdx.x & 7;
    const int bn   = blockIdx.x >> 3;
    const int n    = bn % NCHUNK;
    const int b    = bn / NCHUNK;
    const int h    = hp * 2 + wave;
    WaveLds& L = lw[wave];

    const size_t cbase = ((size_t)(b * SEQ + n * CHUNK)) * D_MODEL + h * DK;
    const size_t head  = (size_t)(b * NCHUNK + n) * N_HEADS + h;

    // TDM: v chunk (64 rows x 64 bf16, row stride D_MODEL) -> L.vbf
    {
        unsigned loff = (unsigned)wave * (unsigned)sizeof(WaveLds)
                      + (unsigned)__builtin_offsetof(WaveLds, vbf);
        tdm_load_tile_bf16(loff, Vb + cbase, CHUNK, DK, D_MODEL);
    }

    // Stage S_prefix (f32 -> bf16) through VALU (conversion needed anyway).
    const float* Sp = Spref + head * (DK * DK);
    for (int i = lane; i < DK * DK; i += 32) L.sbf[i] = f2bf(Sp[i]);

    // z_inter per row: dot(phi_q[row], z_prefix)
    const float* zp = zpref + head * DK;
#pragma unroll
    for (int t = 0; t < 2; ++t) {
        int row = lane + t * 32;
        const unsigned short* qrow = phiQ + cbase + (size_t)row * D_MODEL;
        float s = 0.f;
        for (int k = 0; k < DK; ++k) s += bf2f(qrow[k]) * zp[k];
        L.zrow[row] = s;
    }
    __builtin_amdgcn_s_wait_tensorcnt(0);
    asm volatile("s_wait_dscnt 0" ::: "memory");

    // Phase A: attn = phi_q @ phi_k^T with causal mask; accumulate z_intra.
    for (int it = 0; it < 4; ++it)
        for (int jt = 0; jt < 4; ++jt) {
            if (jt > it) {  // entirely above the diagonal
#pragma unroll
                for (int j = 0; j < 8; ++j)
                    L.attn[(it * 16 + half * 8 + j) * CHUNK + jt * 16 + r] = 0;
                continue;
            }
            CFrag acc;
#pragma unroll
            for (int e = 0; e < 8; ++e) acc.f[e] = 0.f;
#pragma unroll
            for (int cs = 0; cs < 2; ++cs) {
                AFrag a, bb;
                const unsigned short* pa =
                    phiQ + cbase + (size_t)(it * 16 + r) * D_MODEL + cs * 32 + half * 16;
                a.u4[0] = *(const u32x4*)pa;
                a.u4[1] = *(const u32x4*)(pa + 8);
                const unsigned short* pb =
                    phiK + cbase + (size_t)(jt * 16 + r) * D_MODEL + cs * 32 + half * 16;
                bb.u4[0] = *(const u32x4*)pb;
                bb.u4[1] = *(const u32x4*)(pb + 8);
                acc.v = wmma_bf16(a.bf, bb.bf, acc.v);
            }
#pragma unroll
            for (int j = 0; j < 8; ++j) {
                int il = it * 16 + half * 8 + j;
                int jl = jt * 16 + r;
                float f = acc.f[j];
                if (jl > il) f = 0.f;
                L.attn[il * CHUNK + jl] = f2bf(f);
                if (f != 0.f) atomicAdd(&L.zrow[il], f);  // z_intra
            }
        }
    asm volatile("s_wait_dscnt 0" ::: "memory");

    // Phase B: y = (phi_q @ S_prefix + attn @ v) / clip(z)
    for (int it = 0; it < 4; ++it)
        for (int vt = 0; vt < 4; ++vt) {
            CFrag acc;
#pragma unroll
            for (int e = 0; e < 8; ++e) acc.f[e] = 0.f;
#pragma unroll
            for (int cs = 0; cs < 2; ++cs) {  // inter-chunk: K over dk
                AFrag a, bb;
                const unsigned short* pa =
                    phiQ + cbase + (size_t)(it * 16 + r) * D_MODEL + cs * 32 + half * 16;
                a.u4[0] = *(const u32x4*)pa;
                a.u4[1] = *(const u32x4*)(pa + 8);
#pragma unroll
                for (int i = 0; i < 16; ++i)
                    bb.s[i] = L.sbf[(cs * 32 + half * 16 + i) * DK + vt * 16 + r];
                acc.v = wmma_bf16(a.bf, bb.bf, acc.v);
            }
#pragma unroll
            for (int cs = 0; cs < 2; ++cs) {  // intra-chunk: K over j
                AFrag a, bb;
                const unsigned short* pa =
                    &L.attn[(it * 16 + r) * CHUNK + cs * 32 + half * 16];
                a.u4[0] = *(const u32x4*)pa;
                a.u4[1] = *(const u32x4*)(pa + 8);
#pragma unroll
                for (int i = 0; i < 16; ++i)
                    bb.s[i] = L.vbf[(cs * 32 + half * 16 + i) * DK + vt * 16 + r];
                acc.v = wmma_bf16(a.bf, bb.bf, acc.v);
            }
#pragma unroll
            for (int j = 0; j < 8; ++j) {
                int il = it * 16 + half * 8 + j;
                int vc = vt * 16 + r;
                float y = acc.f[j] / fmaxf(L.zrow[il], 1e-6f);
                yout[(size_t)(b * SEQ + n * CHUNK + il) * D_MODEL + h * DK + vc] =
                    f2bf(y);
            }
        }
}

// ---------------------------------------------------------------------------
// Host launcher
// ---------------------------------------------------------------------------
extern "C" void kernel_launch(void* const* d_in, const int* in_sizes, int n_in,
                              void* d_out, int out_size, void* d_ws, size_t ws_size,
                              hipStream_t stream) {
    (void)in_sizes; (void)n_in; (void)out_size; (void)ws_size;
    const float* x  = (const float*)d_in[0];
    const float* Wq = (const float*)d_in[1];
    const float* Wk = (const float*)d_in[2];
    const float* Wv = (const float*)d_in[3];
    const float* Wo = (const float*)d_in[4];

    char* ws = (char*)d_ws;
    size_t off = 0;
    auto alloc = [&](size_t bytes) -> void* {
        void* p = ws + off;
        off += (bytes + 255) & ~(size_t)255;
        return p;
    };
    const size_t TOKBYTES = (size_t)MTOK * D_MODEL * sizeof(unsigned short);
    const size_t WBYTES   = (size_t)D_MODEL * D_MODEL * sizeof(unsigned short);

    unsigned short* x_bf  = (unsigned short*)alloc(TOKBYTES);
    unsigned short* wq_bf = (unsigned short*)alloc(WBYTES);
    unsigned short* wk_bf = (unsigned short*)alloc(WBYTES);
    unsigned short* wv_bf = (unsigned short*)alloc(WBYTES);
    unsigned short* wo_bf = (unsigned short*)alloc(WBYTES);
    unsigned short* q_bf  = (unsigned short*)alloc(TOKBYTES);  // phi_q
    unsigned short* k_bf  = (unsigned short*)alloc(TOKBYTES);  // phi_k
    unsigned short* v_bf  = (unsigned short*)alloc(TOKBYTES);
    unsigned short* y_bf  = (unsigned short*)alloc(TOKBYTES);
    float* KV_ws   = (float*)alloc((size_t)BATCH * NCHUNK * N_HEADS * DK * DK * 4);
    float* Ksum_ws = (float*)alloc((size_t)BATCH * NCHUNK * N_HEADS * DK * 4);

    const int NX = MTOK * D_MODEL;
    const int NW = D_MODEL * D_MODEL;
    k_cast_bf16<<<(NX + 255) / 256, 256, 0, stream>>>(x, x_bf, NX);
    k_cast_bf16<<<(NW + 255) / 256, 256, 0, stream>>>(Wq, wq_bf, NW);
    k_cast_bf16<<<(NW + 255) / 256, 256, 0, stream>>>(Wk, wk_bf, NW);
    k_cast_bf16<<<(NW + 255) / 256, 256, 0, stream>>>(Wv, wv_bf, NW);
    k_cast_bf16<<<(NW + 255) / 256, 256, 0, stream>>>(Wo, wo_bf, NW);

    dim3 ggrid(MTOK / 128, D_MODEL / 128);
    k_gemm_nt<<<ggrid, 256, 0, stream>>>(x_bf, wq_bf, q_bf, nullptr,
                                         MTOK, D_MODEL, D_MODEL, 0);
    k_gemm_nt<<<ggrid, 256, 0, stream>>>(x_bf, wk_bf, k_bf, nullptr,
                                         MTOK, D_MODEL, D_MODEL, 0);
    k_gemm_nt<<<ggrid, 256, 0, stream>>>(x_bf, wv_bf, v_bf, nullptr,
                                         MTOK, D_MODEL, D_MODEL, 1);

    k_chunk_kv<<<BATCH * NCHUNK * (N_HEADS / 4), 128, 0, stream>>>(
        k_bf, v_bf, KV_ws, Ksum_ws);

    int totKV = BATCH * N_HEADS * DK * DK;
    int totZ  = BATCH * N_HEADS * DK;
    k_excl_scan<<<(totKV + 255) / 256, 256, 0, stream>>>(KV_ws, DK * DK);
    k_excl_scan<<<(totZ + 255) / 256, 256, 0, stream>>>(Ksum_ws, DK);

    k_attn<<<BATCH * NCHUNK * (N_HEADS / 2), 64, 0, stream>>>(
        q_bf, k_bf, v_bf, KV_ws, Ksum_ws, y_bf);

    k_gemm_nt<<<ggrid, 256, 0, stream>>>(y_bf, wo_bf, nullptr, (float*)d_out,
                                         MTOK, D_MODEL, D_MODEL, 2);
}
